// EGCN_35948876268088
// MI455X (gfx1250) — 16-bit and 8-bit matrix operand order repaired and checked by reading, compiled, not run on hardware
//
#include <hip/hip_runtime.h>

typedef __attribute__((ext_vector_type(16))) __bf16 v16bf;
typedef __attribute__((ext_vector_type(8)))  float  v8f;
typedef __attribute__((ext_vector_type(8)))  int    v8i;

#define NN 2048
#define DD 256

__global__ void __launch_bounds__(256)
gemm_bf16_nt(const __bf16* __restrict__ A, const __bf16* __restrict__ Bt,
             float* __restrict__ C, int M, int Nc, int K,
             const float* __restrict__ rowScale)
{
  const int lane = threadIdx.x & 31;
  const int wave = threadIdx.x >> 5;
  const int m0 = blockIdx.x * 64  + (wave & 1) * 32;
  const int n0 = blockIdx.y * 128 + (wave >> 1) * 32;
  const int ml = lane & 15;
  const int hi = lane >> 4;
  const __bf16* arow0 = A  + (size_t)(m0 + ml) * K + 16 * hi;
  const __bf16* arow1 = arow0 + (size_t)16 * K;
  const __bf16* brow0 = Bt + (size_t)(n0 + ml) * K + 16 * hi;
  const __bf16* brow1 = brow0 + (size_t)16 * K;
  v8f acc00, acc01, acc10, acc11;
#pragma unroll
  for (int r = 0; r < 8; ++r) { acc00[r] = 0.f; acc01[r] = 0.f; acc10[r] = 0.f; acc11[r] = 0.f; }

  for (int k = 0; k < K; k += 32) {
    if (k + 128 < K) {
      __builtin_prefetch(arow0 + k + 128, 0, 3);
      __builtin_prefetch(arow1 + k + 128, 0, 3);
      __builtin_prefetch(brow0 + k + 128, 0, 3);
      __builtin_prefetch(brow1 + k + 128, 0, 3);
    }
    v16bf a0, a1, b0, b1;
    const __bf16* ap0 = arow0 + k;
    const __bf16* ap1 = arow1 + k;
#pragma unroll
    for (int j = 0; j < 8; ++j) {
      a0[j] = ap0[j]; a0[8 + j] = ap0[8 + j];
      a1[j] = ap1[j]; a1[8 + j] = ap1[8 + j];
    }
    const __bf16* bp0 = brow0 + k;
    const __bf16* bp1 = brow1 + k;
#pragma unroll
    for (int j = 0; j < 16; ++j) { b0[j] = bp0[j]; b1[j] = bp1[j]; }
    acc00 = __builtin_amdgcn_wmma_f32_16x16x32_bf16(false, a0, false, b0, (short)0, acc00, false, false);
    acc01 = __builtin_amdgcn_wmma_f32_16x16x32_bf16(false, a0, false, b1, (short)0, acc01, false, false);
    acc10 = __builtin_amdgcn_wmma_f32_16x16x32_bf16(false, a1, false, b0, (short)0, acc10, false, false);
    acc11 = __builtin_amdgcn_wmma_f32_16x16x32_bf16(false, a1, false, b1, (short)0, acc11, false, false);
  }

#pragma unroll
  for (int r = 0; r < 8; ++r) {
    int r0 = m0 + r + 8 * hi;
    int r1 = r0 + 16;
    float s0 = rowScale ? rowScale[r0] : 1.0f;
    float s1 = rowScale ? rowScale[r1] : 1.0f;
    C[(size_t)r0 * Nc + n0 + ml]      = acc00[r] * s0;
    C[(size_t)r0 * Nc + n0 + 16 + ml] = acc01[r] * s0;
    C[(size_t)r1 * Nc + n0 + ml]      = acc10[r] * s1;
    C[(size_t)r1 * Nc + n0 + 16 + ml] = acc11[r] * s1;
  }
}

__global__ void __launch_bounds__(256)
gemm_iu8_H(const unsigned char* __restrict__ Ed, __bf16* __restrict__ Hout, int n)
{
  const int lane = threadIdx.x & 31;
  const int wave = threadIdx.x >> 5;
  const int m0 = blockIdx.x * 64  + (wave & 1) * 32;
  const int n0 = blockIdx.y * 128 + (wave >> 1) * 32;
  const int ml = lane & 15;
  const int hi = lane >> 4;
  const unsigned char* arow0 = Ed + (size_t)(m0 + ml) * n + 16 * hi;
  const unsigned char* arow1 = arow0 + (size_t)16 * n;
  const unsigned char* brow0 = Ed + (size_t)(n0 + ml) * n + 16 * hi;
  const unsigned char* brow1 = brow0 + (size_t)16 * n;
  const int aoff[8] = {0, 4,  8, 12, 32, 36, 40, 44};
  const int boff[8] = {0, 4,  8, 12, 32, 36, 40, 44};
  v8i acc00, acc01, acc10, acc11;
#pragma unroll
  for (int r = 0; r < 8; ++r) { acc00[r] = 0; acc01[r] = 0; acc10[r] = 0; acc11[r] = 0; }

  for (int k = 0; k < n; k += 64) {
    if (k + 256 < n) {
      __builtin_prefetch(arow0 + k + 256, 0, 3);
      __builtin_prefetch(arow1 + k + 256, 0, 3);
      __builtin_prefetch(brow0 + k + 256, 0, 3);
      __builtin_prefetch(brow1 + k + 256, 0, 3);
    }
    v8i a0, a1, b0, b1;
#pragma unroll
    for (int j = 0; j < 8; ++j) {
      a0[j] = *(const int*)(arow0 + k + aoff[j]);
      a1[j] = *(const int*)(arow1 + k + aoff[j]);
      b0[j] = *(const int*)(brow0 + k + boff[j]);
      b1[j] = *(const int*)(brow1 + k + boff[j]);
    }
    acc00 = __builtin_amdgcn_wmma_i32_16x16x64_iu8(false, a0, false, b0, acc00, false, false);
    acc01 = __builtin_amdgcn_wmma_i32_16x16x64_iu8(false, a0, false, b1, acc01, false, false);
    acc10 = __builtin_amdgcn_wmma_i32_16x16x64_iu8(false, a1, false, b0, acc10, false, false);
    acc11 = __builtin_amdgcn_wmma_i32_16x16x64_iu8(false, a1, false, b1, acc11, false, false);
  }

  const __bf16 one = (__bf16)1.0f, zero = (__bf16)0.0f;
#pragma unroll
  for (int r = 0; r < 8; ++r) {
    int r0 = m0 + r + 8 * hi;
    int r1 = r0 + 16;
    Hout[(size_t)r0 * n + n0 + ml]      = acc00[r] > 0 ? one : zero;
    Hout[(size_t)r0 * n + n0 + 16 + ml] = acc01[r] > 0 ? one : zero;
    Hout[(size_t)r1 * n + n0 + ml]      = acc10[r] > 0 ? one : zero;
    Hout[(size_t)r1 * n + n0 + 16 + ml] = acc11[r] > 0 ? one : zero;
  }
}

__global__ void scale_cvt_bf16(const float* __restrict__ in,
                               const float* __restrict__ rowScale,
                               __bf16* __restrict__ out, int count, int C)
{
  int g = blockIdx.x * 256 + threadIdx.x;
  if (g >= count) return;
  float s = rowScale ? rowScale[g / C] : 1.0f;
  out[g] = (__bf16)(s * in[g]);
}

__global__ void tscale_bf16(const float* __restrict__ in,
                            const float* __restrict__ rowScale,
                            __bf16* __restrict__ out, int M, int C)
{
  int g = blockIdx.x * 256 + threadIdx.x;
  if (g >= M * C) return;
  int i = g / C, c = g % C;
  float s = rowScale ? rowScale[i] : 1.0f;
  out[(size_t)c * M + i] = (__bf16)(s * in[g]);
}

__global__ void row_sq(const float* __restrict__ X, float* __restrict__ sq, int n, int C)
{
  int i = blockIdx.x * 256 + threadIdx.x;
  if (i >= n) return;
  float s = 0.f;
  for (int k = 0; k < C; ++k) { float x = X[(size_t)i * C + k]; s += x * x; }
  sq[i] = s;
}

__global__ void row_abssum(const float* __restrict__ G, const float* __restrict__ sq,
                           float* __restrict__ rowp, int n)
{
  int i = blockIdx.x, t = threadIdx.x;
  float si = sq[i], ts = 0.f;
  for (int j = t; j < n; j += 256)
    ts += fabsf(si + sq[j] - 2.0f * G[(size_t)i * n + j]);
  __shared__ float sm[256];
  sm[t] = ts; __syncthreads();
  for (int s = 128; s > 0; s >>= 1) { if (t < s) sm[t] += sm[t + s]; __syncthreads(); }
  if (t == 0) rowp[i] = sm[0];
}

__global__ void fin_thresh(const float* __restrict__ rowp, float* __restrict__ scal, int n)
{
  int t = threadIdx.x;
  float ts = 0.f;
  for (int i = t; i < n; i += 256) ts += rowp[i];
  __shared__ float sm[256];
  sm[t] = ts; __syncthreads();
  for (int s = 128; s > 0; s >>= 1) { if (t < s) sm[t] += sm[t + s]; __syncthreads(); }
  if (t == 0) scal[0] = sm[0] / ((float)n * (float)n) * 0.2f;
}

__global__ void edge_kernel(const float* __restrict__ G, const float* __restrict__ sq,
                            const float* __restrict__ scal, unsigned char* __restrict__ edge,
                            int n)
{
  float th = scal[0];
  size_t base = ((size_t)blockIdx.x * 256 + threadIdx.x) * 8;
#pragma unroll
  for (int e = 0; e < 8; ++e) {
    size_t idx = base + e;
    int i = (int)(idx >> 11), j = (int)(idx & (NN - 1));
    edge[idx] = (fabsf(sq[i] + sq[j] - 2.0f * G[idx]) < th) ? 1 : 0;
  }
}

__global__ void row_dv(const __bf16* __restrict__ H, float* __restrict__ dvinv, int n)
{
  int i = blockIdx.x, t = threadIdx.x;
  float ts = 0.f;
  for (int j = t; j < n; j += 256) ts += (float)H[(size_t)i * n + j];
  __shared__ float sm[256];
  sm[t] = ts; __syncthreads();
  for (int s = 128; s > 0; s >>= 1) { if (t < s) sm[t] += sm[t + s]; __syncthreads(); }
  if (t == 0) dvinv[i] = rsqrtf(sm[0]);
}

__global__ void gemm_skinny(const float* __restrict__ A, const float* __restrict__ W,
                            float* __restrict__ out, int n)
{
  int g = blockIdx.x * 256 + threadIdx.x;
  if (g >= n * 8) return;
  int i = g >> 3, c = g & 7;
  float s = 0.f;
  for (int k = 0; k < DD; ++k) s += A[(size_t)i * DD + k] * W[k * 8 + c];
  out[g] = s;
}

__global__ void dotuv(const float* __restrict__ WQ, const float* __restrict__ WK,
                      const float* __restrict__ al, float* __restrict__ u,
                      float* __restrict__ v, int n)
{
  int i = blockIdx.x * 256 + threadIdx.x;
  if (i >= n) return;
  float su = 0.f, sv = 0.f;
#pragma unroll
  for (int c = 0; c < 8; ++c) {
    su += WQ[i * 8 + c] * al[c];
    sv += WK[i * 8 + c] * al[8 + c];
  }
  u[i] = su; v[i] = sv;
}

__device__ __forceinline__ float lrelu(float x) { return x > 0.f ? x : 0.2f * x; }

__global__ void attn_stats(const __bf16* __restrict__ H, const float* __restrict__ u,
                           const float* __restrict__ v, float* __restrict__ mrow,
                           float* __restrict__ sinv, int n)
{
  int i = blockIdx.x, t = threadIdx.x;
  const __bf16* hr = H + (size_t)i * n;
  float ui = u[i], tm = -3.4e38f;
  for (int j = t; j < n; j += 256)
    if ((float)hr[j] > 0.5f) tm = fmaxf(tm, lrelu(ui + v[j]));
  __shared__ float sm[256];
  sm[t] = tm; __syncthreads();
  for (int s = 128; s > 0; s >>= 1) { if (t < s) sm[t] = fmaxf(sm[t], sm[t + s]); __syncthreads(); }
  float m = sm[0]; __syncthreads();
  float ts = 0.f;
  for (int j = t; j < n; j += 256)
    if ((float)hr[j] > 0.5f) ts += __expf(lrelu(ui + v[j]) - m);
  sm[t] = ts; __syncthreads();
  for (int s = 128; s > 0; s >>= 1) { if (t < s) sm[t] += sm[t + s]; __syncthreads(); }
  if (t == 0) { mrow[i] = m; sinv[i] = 1.0f / sm[0]; }
}

__global__ void attn_accum(const __bf16* __restrict__ H, const float* __restrict__ u,
                           const float* __restrict__ v, const float* __restrict__ mrow,
                           const float* __restrict__ sinv, const float* __restrict__ agg,
                           float* __restrict__ out, int ostride, int ooff, int n)
{
  int j = blockIdx.x, t = threadIdx.x;
  const __bf16* hr = H + (size_t)j * n;
  float vj = v[j];
  float acc[8];
#pragma unroll
  for (int c = 0; c < 8; ++c) acc[c] = 0.f;
  for (int i = t; i < n; i += 256) {
    if ((float)hr[i] > 0.5f) {
      float w = __expf(lrelu(u[i] + vj) - mrow[i]) * sinv[i];
      const float* ag = agg + (size_t)i * 8;
#pragma unroll
      for (int c = 0; c < 8; ++c) acc[c] += w * ag[c];
    }
  }
  __shared__ float red[8 * 256];
#pragma unroll
  for (int c = 0; c < 8; ++c) red[c * 256 + t] = acc[c];
  __syncthreads();
  for (int s = 128; s > 0; s >>= 1) {
    if (t < s) {
#pragma unroll
      for (int c = 0; c < 8; ++c) red[c * 256 + t] += red[c * 256 + t + s];
    }
    __syncthreads();
  }
  if (t < 8) {
    float o = red[t * 256];
    out[(size_t)j * ostride + ooff + t] = o > 0.f ? o : (__expf(o) - 1.f);
  }
}

__global__ void x2_kernel(const float* __restrict__ out1, const float* __restrict__ lw,
                          const float* __restrict__ lb, float* __restrict__ X2)
{
  int g = blockIdx.x * 256 + threadIdx.x;
  int i = g >> 8, c = g & 255;
  float s = lb[c];
#pragma unroll
  for (int k = 0; k < 32; ++k) s += out1[i * 32 + k] * lw[c * 32 + k];
  X2[g] = s > 0.f ? s : (__expf(s) - 1.f);
}

__device__ void gru_run(float* Q, float* R, float* Zb, float* Hb,
                        const float* W, const float* U, int cols, int t)
{
  int i = t / (cols > 0 ? cols : 1), j = t % (cols > 0 ? cols : 1);
  bool act = t < 16 * cols;
  for (int step = 0; step < 3; ++step) {
    if (act) {
      float z = 0.f, r = 0.f;
      for (int k = 0; k < 16; ++k) {
        float q = Q[k * cols + j];
        z += (W[i * 16 + k] + U[i * 16 + k]) * q;
        r += (W[256 + i * 16 + k] + U[256 + i * 16 + k]) * q;
      }
      z = 1.f / (1.f + __expf(-z));
      r = 1.f / (1.f + __expf(-r));
      Zb[i * cols + j] = z;
      R[i * cols + j] = r * Q[i * cols + j];
    }
    __syncthreads();
    if (act) {
      float h = 0.f;
      for (int k = 0; k < 16; ++k)
        h += W[512 + i * 16 + k] * Q[k * cols + j] +
             U[512 + i * 16 + k] * R[k * cols + j];
      h = tanhf(h);
      float z = Zb[i * cols + j];
      Hb[i * cols + j] = (1.f - z) * Q[i * cols + j] + z * h;
    }
    __syncthreads();
    if (act) Q[i * cols + j] = Hb[i * cols + j];
    __syncthreads();
  }
}

__global__ void gru_prep(const float* __restrict__ A1, const float* __restrict__ W1,
                         const float* __restrict__ U1, const float* __restrict__ A2,
                         const float* __restrict__ W2, const float* __restrict__ U2,
                         float* __restrict__ ax1, float* __restrict__ ae1,
                         float* __restrict__ a2)
{
  __shared__ float Q[128], R[128], Zb[128], Hb[128];
  int t = threadIdx.x;
  if (t < 128) Q[t] = A1[t];
  __syncthreads();
  gru_run(Q, R, Zb, Hb, W1, U1, 8, t);
  if (t < 64) { int h = t >> 4, row = t & 15; ax1[t] = Q[row * 8 + h]; ae1[t] = Q[row * 8 + 4 + h]; }
  __syncthreads();
  if (t < 32) Q[t] = A2[t];
  __syncthreads();
  gru_run(Q, R, Zb, Hb, W2, U2, 2, t);
  if (t < 16) { a2[t] = Q[t * 2 + 0]; a2[16 + t] = Q[t * 2 + 1]; }
}

extern "C" void kernel_launch(void* const* d_in, const int* in_sizes, int n_in,
                              void* d_out, int out_size, void* d_ws, size_t ws_size,
                              hipStream_t stream)
{
  (void)in_sizes; (void)n_in; (void)out_size; (void)ws_size;
  const float* X     = (const float*)d_in[0] + (size_t)2 * NN * DD;
  const float* theta = (const float*)d_in[1];
  const float* lin_w = (const float*)d_in[2];
  const float* lin_b = (const float*)d_in[3];
  const float* Watt1 = (const float*)d_in[4];
  const float* A1    = (const float*)d_in[5];
  const float* g1W   = (const float*)d_in[6];
  const float* g1U   = (const float*)d_in[7];
  const float* Watt2 = (const float*)d_in[8];
  const float* A2    = (const float*)d_in[9];
  const float* g2W   = (const float*)d_in[10];
  const float* g2U   = (const float*)d_in[11];
  float* outp = (float*)d_out;

  const size_t MB = 1024ull * 1024ull;
  char* w = (char*)d_ws;
  float*  G     = (float*)(w + 0);
  float*  S     = (float*)(w + 0);
  float*  Ef    = (float*)(w + 2 * MB);
  float*  XO    = (float*)(w + 4 * MB);
  float*  X2    = (float*)(w + 6 * MB);
  float*  OUT1  = (float*)(w + 8 * MB);
  float*  P     = (float*)(w + 8 * MB + 512 * 1024);
  float*  QE    = (float*)(w + 8 * MB + 576 * 1024);
  float*  ENEW  = (float*)(w + 8 * MB + 640 * 1024);
  float*  U     = (float*)(w + 8 * MB + 704 * 1024);
  float*  V     = (float*)(w + 8 * MB + 712 * 1024);
  float*  MR    = (float*)(w + 8 * MB + 720 * 1024);
  float*  SI    = (float*)(w + 8 * MB + 728 * 1024);
  __bf16* Hbf   = (__bf16*)(w + 16 * MB);
  unsigned char* EDG = (unsigned char*)(w + 24 * MB);
  __bf16* Xh    = (__bf16*)(w + 28 * MB);
  __bf16* Yt    = (__bf16*)(w + 29 * MB);
  __bf16* Th    = (__bf16*)(w + 30 * MB);
  __bf16* Zt    = (__bf16*)(w + 31 * MB);
  __bf16* ThT   = (__bf16*)(w + 32 * MB);
  float*  SQ    = (float*)(w + 32 * MB + 128 * 1024);
  float*  DV    = (float*)(w + 32 * MB + 136 * 1024);
  float*  ROWP  = (float*)(w + 32 * MB + 144 * 1024);
  float*  SCAL  = (float*)(w + 32 * MB + 152 * 1024);
  float*  AX1   = (float*)(w + 32 * MB + 156 * 1024);
  float*  AE1   = (float*)(w + 32 * MB + 157 * 1024);
  float*  A2F   = (float*)(w + 32 * MB + 158 * 1024);

  const int TPB = 256;
  scale_cvt_bf16<<<NN * DD / TPB, TPB, 0, stream>>>(X, nullptr, Xh, NN * DD, DD);
  row_sq<<<NN / TPB, TPB, 0, stream>>>(X, SQ, NN, DD);
  gemm_bf16_nt<<<dim3(NN / 64, NN / 128), TPB, 0, stream>>>(Xh, Xh, G, NN, NN, DD, nullptr);
  row_abssum<<<NN, TPB, 0, stream>>>(G, SQ, ROWP, NN);
  fin_thresh<<<1, TPB, 0, stream>>>(ROWP, SCAL, NN);
  edge_kernel<<<NN * NN / (TPB * 8), TPB, 0, stream>>>(G, SQ, SCAL, EDG, NN);
  gemm_iu8_H<<<dim3(NN / 64, NN / 128), TPB, 0, stream>>>(EDG, Hbf, NN);
  row_dv<<<NN, TPB, 0, stream>>>(Hbf, DV, NN);
  tscale_bf16<<<NN * DD / TPB, TPB, 0, stream>>>(X, DV, Yt, NN, DD);
  gemm_bf16_nt<<<dim3(NN / 64, DD / 128), TPB, 0, stream>>>(Hbf, Yt, S, NN, DD, NN, nullptr);
  scale_cvt_bf16<<<NN * DD / TPB, TPB, 0, stream>>>(S, DV, Th, NN * DD, DD);
  tscale_bf16<<<DD * DD / TPB, TPB, 0, stream>>>(theta, nullptr, ThT, DD, DD);
  gemm_bf16_nt<<<dim3(NN / 64, DD / 128), TPB, 0, stream>>>(Th, ThT, Ef, NN, DD, DD, nullptr);
  tscale_bf16<<<NN * DD / TPB, TPB, 0, stream>>>(Ef, DV, Zt, NN, DD);
  gemm_bf16_nt<<<dim3(NN / 64, DD / 128), TPB, 0, stream>>>(Hbf, Zt, XO, NN, DD, NN, DV);

  gru_prep<<<1, TPB, 0, stream>>>(A1, g1W, g1U, A2, g2W, g2U, AX1, AE1, A2F);

  auto head = [&](const float* Qm, const float* Km, const float* Wh,
                  const float* ax, const float* ae, float* op, int ostr, int ooff) {
    gemm_skinny<<<NN * 8 / TPB, TPB, 0, stream>>>(Qm, Wh, P, NN);
    gemm_skinny<<<NN * 8 / TPB, TPB, 0, stream>>>(Km, Wh, QE, NN);
    dotuv<<<NN / TPB, TPB, 0, stream>>>(P, QE, ax, U, V, NN);
    attn_stats<<<NN, TPB, 0, stream>>>(Hbf, U, V, MR, SI, NN);
    attn_accum<<<NN, TPB, 0, stream>>>(Hbf, U, V, MR, SI, P, ENEW, 8, 0, NN);
    dotuv<<<NN / TPB, TPB, 0, stream>>>(QE, P, ae, U, V, NN);
    attn_stats<<<NN, TPB, 0, stream>>>(Hbf, U, V, MR, SI, NN);
    attn_accum<<<NN, TPB, 0, stream>>>(Hbf, U, V, MR, SI, ENEW, op, ostr, ooff, NN);
  };
  for (int h = 0; h < 4; ++h)
    head(XO, Ef, Watt1 + (size_t)h * DD * 8, AX1 + h * 16, AE1 + h * 16, OUT1, 32, h * 8);

  x2_kernel<<<NN * DD / TPB, TPB, 0, stream>>>(OUT1, lin_w, lin_b, X2);
  head(X2, Ef, Watt2, A2F, A2F + 16, outp, 8, 0);
}
